// Block_18528488915394
// MI455X (gfx1250) — matrix operand order repaired and checked by reading, not compiled, not run on hardware
//
#include <hip/hip_runtime.h>
#include <cstdint>
#include <cstddef>

// ---------------------------------------------------------------------------
// Problem constants (match reference)
// ---------------------------------------------------------------------------
constexpr int  Bb   = 2;
constexpr int  Tt   = 2048;
constexpr int  Cc   = 2048;
constexpr int  NH   = 16;
constexpr int  Gg   = 4;       // KV groups
constexpr int  Dd   = 128;     // head size
constexpr int  HID  = 4 * Cc;  // 8192
constexpr int  NKV  = (NH + 2 * Gg) * Dd;  // 3072
constexpr int  M4   = Bb * Tt; // 4096 rows

typedef __bf16 bf16;
typedef __attribute__((ext_vector_type(16))) __bf16 v16bf;
typedef __attribute__((ext_vector_type(8)))  float  v8f;

__device__ __forceinline__ bf16 f2bf(float f) {
  uint32_t u = __builtin_bit_cast(uint32_t, f);
  uint32_t r = u + 0x7FFFu + ((u >> 16) & 1u);   // round-to-nearest-even
  uint16_t h = (uint16_t)(r >> 16);
  return __builtin_bit_cast(bf16, h);
}

// ---------------------------------------------------------------------------
// CDNA5 async global->LDS copy (ASYNCcnt) with portable fallback
// ---------------------------------------------------------------------------
#if defined(__has_builtin)
#if __has_builtin(__builtin_amdgcn_global_load_async_to_lds_b128) && \
    __has_builtin(__builtin_amdgcn_s_wait_asynccnt)
#define USE_ASYNC_LDS 1
#endif
#endif

typedef int async_v4i __attribute__((vector_size(16)));
typedef __attribute__((address_space(1))) async_v4i* gbl_v4i_ptr;
typedef __attribute__((address_space(3))) async_v4i* lds_v4i_ptr;

__device__ __forceinline__ void async_copy_b128(const bf16* g, bf16* l) {
#ifdef USE_ASYNC_LDS
  __builtin_amdgcn_global_load_async_to_lds_b128(
      (gbl_v4i_ptr)g, (lds_v4i_ptr)l, 0, 0);
#else
  *(uint4*)l = *(const uint4*)g;
#endif
}
__device__ __forceinline__ void async_wait() {
#ifdef USE_ASYNC_LDS
  __builtin_amdgcn_s_wait_asynccnt(0);
#endif
}

// Fragment gather: two contiguous 16B runs -> one v16bf (forces ds_load_b128)
__device__ __forceinline__ v16bf load_frag2(const bf16* p0, const bf16* p1) {
  union { uint4 q[2]; v16bf v; } u;
  u.q[0] = *(const uint4*)p0;
  u.q[1] = *(const uint4*)p1;
  return u.v;
}

// ---------------------------------------------------------------------------
// fp32 -> bf16 bulk convert (weights)
// ---------------------------------------------------------------------------
__global__ __launch_bounds__(256) void cvt_bf16_kernel(
    const float* __restrict__ in, bf16* __restrict__ out, size_t n) {
  size_t i = (size_t)blockIdx.x * 256 + threadIdx.x;
  if (i < n) out[i] = f2bf(in[i]);
}

// ---------------------------------------------------------------------------
// LayerNorm over C, bf16 output
// ---------------------------------------------------------------------------
__global__ __launch_bounds__(256) void ln_bf16_kernel(
    const float* __restrict__ x, const float* __restrict__ w,
    const float* __restrict__ bias, bf16* __restrict__ out) {
  const int row = blockIdx.x;
  const float* xr = x + (size_t)row * Cc;
  __shared__ float red0[256], red1[256];
  float s = 0.f, s2 = 0.f;
  for (int i = threadIdx.x; i < Cc; i += 256) {
    float v = xr[i]; s += v; s2 += v * v;
  }
  red0[threadIdx.x] = s; red1[threadIdx.x] = s2;
  __syncthreads();
  for (int off = 128; off > 0; off >>= 1) {
    if (threadIdx.x < (unsigned)off) {
      red0[threadIdx.x] += red0[threadIdx.x + off];
      red1[threadIdx.x] += red1[threadIdx.x + off];
    }
    __syncthreads();
  }
  const float mean = red0[0] * (1.0f / Cc);
  const float var  = red1[0] * (1.0f / Cc) - mean * mean;
  const float inv  = rsqrtf(var + 1e-5f);
  for (int i = threadIdx.x; i < Cc; i += 256) {
    float v = (xr[i] - mean) * inv * w[i] + bias[i];
    out[(size_t)row * Cc + i] = f2bf(v);
  }
}

// ---------------------------------------------------------------------------
// WMMA bf16 GEMM: out = A[M,K](bf16) @ W[K,N](bf16) (+residual) (gelu?)
// 128x128x32 block tile, 8 waves -> 32x64 per wave (2x4 WMMA accumulators).
// Double-buffered LDS; B tile stored transposed so B-fragments are contiguous.
// ---------------------------------------------------------------------------
#define GBM 128
#define GBN 128
#define GBK 32
#define LDK 40     // GBK + 8 -> 80B row stride (16B aligned)

__global__ __launch_bounds__(256) void gemm_bf16_kernel(
    const bf16* __restrict__ A, const bf16* __restrict__ Bw,
    const float* __restrict__ residual,
    float* __restrict__ outF, bf16* __restrict__ outB,
    int M, int N, int K, int doGelu) {
  __shared__ __align__(16) bf16 As[2][GBM * LDK];   // [row][k]
  __shared__ __align__(16) bf16 Bt[2][GBN * LDK];   // [col][k]  (transposed)

  const int tid   = threadIdx.x;
  const int wave  = tid >> 5;
  const int lane  = tid & 31;
  const int wm    = wave & 3;    // 4 x 32-row slabs
  const int wn    = wave >> 2;   // 2 x 64-col slabs
  const int nlane = lane & 15;
  const int hi    = lane >> 4;

  const int bm0 = blockIdx.y * GBM;
  const int bn0 = blockIdx.x * GBN;

  v8f acc[2][4];
#pragma unroll
  for (int i = 0; i < 2; ++i)
#pragma unroll
    for (int j = 0; j < 4; ++j)
#pragma unroll
      for (int r = 0; r < 8; ++r) acc[i][j][r] = 0.f;

  const int arow = tid >> 1, acol = (tid & 1) * 16;   // A tile 128x32
  const int brow = tid >> 3, bcol = (tid & 7) * 16;   // B tile 32x128

  auto loadA = [&](int buf, int k0) {
    const bf16* g = A + (size_t)(bm0 + arow) * K + k0 + acol;
    bf16* l = &As[buf][arow * LDK + acol];
    async_copy_b128(g, l);
    async_copy_b128(g + 8, l + 8);
  };
  auto loadB = [&](int buf, int k0) {  // transpose in registers
    const uint4* src = (const uint4*)(Bw + (size_t)(k0 + brow) * N + bn0 + bcol);
    union { uint4 q[2]; bf16 h[16]; } tb;
    tb.q[0] = src[0]; tb.q[1] = src[1];
#pragma unroll
    for (int i = 0; i < 16; ++i) Bt[buf][(bcol + i) * LDK + brow] = tb.h[i];
  };

  loadA(0, 0);
  loadB(0, 0);
  async_wait();
  __syncthreads();

  int cur = 0;
  for (int k0 = 0; k0 < K; k0 += GBK) {
    const int nk = k0 + GBK;
    if (nk < K) { loadA(cur ^ 1, nk); loadB(cur ^ 1, nk); }
    if (nk + GBK < K) {
      __builtin_prefetch(A + (size_t)(bm0 + arow) * K + nk + GBK + acol, 0, 1);
      __builtin_prefetch(Bw + (size_t)(nk + GBK + brow) * N + bn0 + bcol, 0, 1);
    }

    v16bf afr[2], bfr[4];
#pragma unroll
    for (int mt = 0; mt < 2; ++mt) {
      // A layout: lanes 0-15 K={0..7,16..23}, lanes 16-31 K={8..15,24..31}
      const bf16* p = &As[cur][(wm * 32 + mt * 16 + nlane) * LDK + hi * 8];
      afr[mt] = load_frag2(p, p + 16);
    }
#pragma unroll
    for (int nt = 0; nt < 4; ++nt) {
      // B layout: lanes 0-15 K={0..7,16..23}, lanes 16-31 K={8..15,24..31}
      const bf16* p = &Bt[cur][(wn * 64 + nt * 16 + nlane) * LDK + hi * 8];
      bfr[nt] = load_frag2(p, p + 16);
    }
#pragma unroll
    for (int mt = 0; mt < 2; ++mt)
#pragma unroll
      for (int nt = 0; nt < 4; ++nt)
        acc[mt][nt] = __builtin_amdgcn_wmma_f32_16x16x32_bf16(
            false, afr[mt], false, bfr[nt], (short)0, acc[mt][nt], false, false);

    async_wait();
    __syncthreads();
    cur ^= 1;
  }

  // Epilogue: C/D layout m = r + 8*hi, n = nlane
  const bool hasRes = residual != nullptr;
  const bool hasF   = outF != nullptr;
  const bool hasB   = outB != nullptr;
#pragma unroll
  for (int mt = 0; mt < 2; ++mt)
#pragma unroll
    for (int nt = 0; nt < 4; ++nt)
#pragma unroll
      for (int r = 0; r < 8; ++r) {
        const int row = bm0 + wm * 32 + mt * 16 + r + 8 * hi;
        const int col = bn0 + wn * 64 + nt * 16 + nlane;
        float v = acc[mt][nt][r];
        if (doGelu) v = 0.5f * v * (1.0f + erff(v * 0.7071067811865475f));
        if (hasRes) v += residual[(size_t)row * N + col];
        if (hasF) outF[(size_t)row * N + col] = v;
        if (hasB) outB[(size_t)row * N + col] = f2bf(v);
      }
}

// ---------------------------------------------------------------------------
// RoPE + QKV split/reorder. qkv fp32 [B,T,G,6,D] ->
//   q bf16 [B,NH,T,D] (pre-scaled by 1/sqrt(D)), k/v bf16 [B,G,T,D]
// ---------------------------------------------------------------------------
__global__ __launch_bounds__(128) void rope_split_kernel(
    const float* __restrict__ qkv, const float* __restrict__ cosT,
    const float* __restrict__ sinT, bf16* __restrict__ qo,
    bf16* __restrict__ ko, bf16* __restrict__ vo) {
  const int d    = threadIdx.x;    // 0..127
  const int slot = blockIdx.y;     // 0..23  = g*6 + j
  const int bt   = blockIdx.x;     // 0..B*T-1
  const int b = bt / Tt, t = bt % Tt;
  const int g = slot / 6, j = slot % 6;
  const float* src = qkv + (size_t)bt * NKV + (size_t)slot * Dd;
  float v = src[d];
  if (j <= 4) {  // rope on q (j<4) and k (j==4)
    const float other = (d < Dd / 2) ? -src[d + Dd / 2] : src[d - Dd / 2];
    v = v * cosT[t * Dd + d] + other * sinT[t * Dd + d];
  }
  if (j < 4) {
    const int h = g * 4 + j;
    qo[(((size_t)b * NH + h) * Tt + t) * Dd + d] = f2bf(v * 0.08838834764831845f);
  } else if (j == 4) {
    ko[(((size_t)b * Gg + g) * Tt + t) * Dd + d] = f2bf(v);
  } else {
    vo[(((size_t)b * Gg + g) * Tt + t) * Dd + d] = f2bf(v);
  }
}

// ---------------------------------------------------------------------------
// Flash attention (causal, GQA). 8 waves/block; each wave owns 16 query rows
// (block covers 128). KV streamed in 32-row tiles through LDS. V tile stored
// transposed so P@V B-fragments are contiguous. All fragments = 2x b128.
// y written as bf16 [B,T,NH,D].
// ---------------------------------------------------------------------------
#define FA_KSTR 136  // K tile row stride: 128+8 (272B, 16B aligned)
#define FA_VSTR 40   // Vt row stride: 32+8 (80B, 16B aligned)
#define FA_PSTR 40   // P scratch row stride (80B, 16B aligned)

__global__ __launch_bounds__(256) void flash_attn_kernel(
    const bf16* __restrict__ qg, const bf16* __restrict__ kg,
    const bf16* __restrict__ vg, bf16* __restrict__ yg) {
  __shared__ __align__(16) bf16 Ks[32 * FA_KSTR];        // [key][d]
  __shared__ __align__(16) bf16 Vt[Dd * FA_VSTR];        // [d][key] (transposed)
  __shared__ __align__(16) bf16 Ps[8 * 16 * FA_PSTR];    // per-wave P scratch

  const int tid   = threadIdx.x;
  const int wave  = tid >> 5;
  const int lane  = tid & 31;
  const int nlane = lane & 15;
  const int hi    = lane >> 4;

  const int qblk = blockIdx.x;
  const int h    = blockIdx.y;
  const int b    = blockIdx.z;
  const int g    = h >> 2;  // Q_PER_KV = 4

  const int q0 = qblk * 128 + wave * 16;

  const bf16* qp = qg + (((size_t)b * NH + h) * Tt) * Dd;
  const bf16* kp = kg + (((size_t)b * Gg + g) * Tt) * Dd;
  const bf16* vp = vg + (((size_t)b * Gg + g) * Tt) * Dd;

  // Q A-fragments for the 4 K-steps over d (d = 32*kk .. 32*kk+31)
  v16bf aq[4];
  {
    const bf16* qrow = qp + (size_t)(q0 + nlane) * Dd;
#pragma unroll
    for (int kk = 0; kk < 4; ++kk)
      aq[kk] = load_frag2(qrow + 32 * kk + hi * 8,
                          qrow + 32 * kk + 16 + hi * 8);
  }

  v8f o[8];
#pragma unroll
  for (int t = 0; t < 8; ++t)
#pragma unroll
    for (int r = 0; r < 8; ++r) o[t][r] = 0.f;
  float mrow[8], lrow[8];
#pragma unroll
  for (int r = 0; r < 8; ++r) { mrow[r] = -1e30f; lrow[r] = 0.f; }

  const int kv_end = qblk * 128 + 128;  // causal bound for this block
  const int ldrow = tid >> 3, ldcol = (tid & 7) * 16;

  for (int kv = 0; kv < kv_end; kv += 32) {
    {  // K tile: async b128 into [key][d]
      const bf16* gk = kp + (size_t)(kv + ldrow) * Dd + ldcol;
      bf16* lk = Ks + ldrow * FA_KSTR + ldcol;
      async_copy_b128(gk, lk);
      async_copy_b128(gk + 8, lk + 8);
    }
    {  // V tile: register transpose into [d][key]
      const uint4* vs = (const uint4*)(vp + (size_t)(kv + ldrow) * Dd + ldcol);
      union { uint4 q[2]; bf16 h[16]; } tv;
      tv.q[0] = vs[0]; tv.q[1] = vs[1];
#pragma unroll
      for (int i = 0; i < 16; ++i) Vt[(ldcol + i) * FA_VSTR + ldrow] = tv.h[i];
    }
    async_wait();
    __syncthreads();

    // ---- S = Q @ K^T : 16 x 32 (two 16x16 accumulators) ----
    v8f s0, s1;
#pragma unroll
    for (int r = 0; r < 8; ++r) { s0[r] = 0.f; s1[r] = 0.f; }
#pragma unroll
    for (int kk = 0; kk < 4; ++kk) {
      const bf16* kb0 = Ks + nlane * FA_KSTR + 32 * kk + hi * 8;
      const bf16* kb1 = Ks + (16 + nlane) * FA_KSTR + 32 * kk + hi * 8;
      v16bf b0 = load_frag2(kb0, kb0 + 16);
      v16bf b1 = load_frag2(kb1, kb1 + 16);
      s0 = __builtin_amdgcn_wmma_f32_16x16x32_bf16(false, aq[kk], false, b0,
                                                   (short)0, s0, false, false);
      s1 = __builtin_amdgcn_wmma_f32_16x16x32_bf16(false, aq[kk], false, b1,
                                                   (short)0, s1, false, false);
    }

    // ---- causal mask + online softmax (accumulator layout: m=r+8*hi, n=nlane)
#pragma unroll
    for (int r = 0; r < 8; ++r) {
      const int qidx = q0 + r + 8 * hi;
      float a  = (kv + nlane      > qidx) ? -1e30f : s0[r];
      float bb = (kv + 16 + nlane > qidx) ? -1e30f : s1[r];
      float mx = fmaxf(a, bb);
#pragma unroll
      for (int off = 1; off < 16; off <<= 1) mx = fmaxf(mx, __shfl_xor(mx, off));
      const float nm = fmaxf(mrow[r], mx);
      const float sc = __expf(mrow[r] - nm);
      mrow[r] = nm;
      const float p0 = __expf(a - nm), p1 = __expf(bb - nm);
      float rs = p0 + p1;
#pragma unroll
      for (int off = 1; off < 16; off <<= 1) rs += __shfl_xor(rs, off);
      lrow[r] = lrow[r] * sc + rs;
#pragma unroll
      for (int t = 0; t < 8; ++t) o[t][r] *= sc;
      bf16* pr = Ps + (wave * 16 + r + 8 * hi) * FA_PSTR;
      pr[nlane]      = f2bf(p0);
      pr[16 + nlane] = f2bf(p1);
    }

    // ---- O += P @ V  (WMMA K = 32 keys; 8 d-tiles of 16) ----
    v16bf ap;
    {
      const bf16* pl = Ps + (wave * 16 + nlane) * FA_PSTR + hi * 8;
      ap = load_frag2(pl, pl + 16);
    }
#pragma unroll
    for (int t = 0; t < 8; ++t) {
      const bf16* pv = Vt + (t * 16 + nlane) * FA_VSTR + hi * 8;
      v16bf bv = load_frag2(pv, pv + 16);
      o[t] = __builtin_amdgcn_wmma_f32_16x16x32_bf16(false, ap, false, bv,
                                                     (short)0, o[t], false, false);
    }
    __syncthreads();
  }

  // normalize + store y as [B, T, NH, D] bf16
#pragma unroll
  for (int r = 0; r < 8; ++r) {
    const float inv = 1.0f / lrow[r];
    const int trow = q0 + r + 8 * hi;
    bf16* yr = yg + (((size_t)(b * Tt + trow) * NH) + h) * Dd;
#pragma unroll
    for (int t = 0; t < 8; ++t)
      yr[t * 16 + nlane] = f2bf(o[t][r] * inv);
  }
}

// ---------------------------------------------------------------------------
// Host-side orchestration
// ---------------------------------------------------------------------------
extern "C" void kernel_launch(void* const* d_in, const int* in_sizes, int n_in,
                              void* d_out, int out_size, void* d_ws, size_t ws_size,
                              hipStream_t stream) {
  (void)in_sizes; (void)n_in; (void)out_size; (void)ws_size;
  const float* x      = (const float*)d_in[0];
  const float* cosT   = (const float*)d_in[1];
  const float* sinT   = (const float*)d_in[2];
  const float* ln1_w  = (const float*)d_in[3];
  const float* ln1_b  = (const float*)d_in[4];
  const float* ln2_w  = (const float*)d_in[5];
  const float* ln2_b  = (const float*)d_in[6];
  const float* w_attn = (const float*)d_in[7];
  const float* w_proj = (const float*)d_in[8];
  const float* w_fc   = (const float*)d_in[9];
  const float* w_mlp  = (const float*)d_in[10];
  float* out = (float*)d_out;

  // Workspace carve-up (256B aligned chunks)
  char* ws = (char*)d_ws;
  size_t off = 0;
  auto alloc = [&](size_t bytes) {
    char* p = ws + off;
    off += (bytes + 255) & ~(size_t)255;
    return p;
  };
  bf16*  wbf_attn = (bf16*)alloc((size_t)Cc * NKV * 2);
  bf16*  wbf_proj = (bf16*)alloc((size_t)Cc * Cc * 2);
  bf16*  wbf_fc   = (bf16*)alloc((size_t)Cc * HID * 2);
  bf16*  wbf_mlp  = (bf16*)alloc((size_t)HID * Cc * 2);
  bf16*  n1bf     = (bf16*)alloc((size_t)M4 * Cc * 2);
  float* qkv      = (float*)alloc((size_t)M4 * NKV * 4);
  bf16*  qbf      = (bf16*)alloc((size_t)Bb * NH * Tt * Dd * 2);
  bf16*  kbf      = (bf16*)alloc((size_t)Bb * Gg * Tt * Dd * 2);
  bf16*  vbf      = (bf16*)alloc((size_t)Bb * Gg * Tt * Dd * 2);
  bf16*  ybf      = (bf16*)alloc((size_t)M4 * Cc * 2);
  float* x2       = (float*)alloc((size_t)M4 * Cc * 4);
  bf16*  n2bf     = (bf16*)alloc((size_t)M4 * Cc * 2);
  bf16*  actbf    = (bf16*)alloc((size_t)M4 * HID * 2);

  auto cvt = [&](const float* src, bf16* dst, size_t n) {
    cvt_bf16_kernel<<<dim3((unsigned)((n + 255) / 256)), 256, 0, stream>>>(src, dst, n);
  };
  cvt(w_attn, wbf_attn, (size_t)Cc * NKV);
  cvt(w_proj, wbf_proj, (size_t)Cc * Cc);
  cvt(w_fc,   wbf_fc,   (size_t)Cc * HID);
  cvt(w_mlp,  wbf_mlp,  (size_t)HID * Cc);

  // LN1
  ln_bf16_kernel<<<M4, 256, 0, stream>>>(x, ln1_w, ln1_b, n1bf);

  // QKV = n1 @ w_attn  -> fp32
  gemm_bf16_kernel<<<dim3(NKV / GBN, M4 / GBM), 256, 0, stream>>>(
      n1bf, wbf_attn, nullptr, qkv, nullptr, M4, NKV, Cc, 0);

  // RoPE + split (+1/sqrt(D) folded into q)
  rope_split_kernel<<<dim3(Bb * Tt, (Gg * 6)), 128, 0, stream>>>(
      qkv, cosT, sinT, qbf, kbf, vbf);

  // Flash attention -> ybf [B,T,NH,D]
  flash_attn_kernel<<<dim3(Tt / 128, NH, Bb), 256, 0, stream>>>(qbf, kbf, vbf, ybf);

  // x2 = x + y @ w_proj  -> fp32
  gemm_bf16_kernel<<<dim3(Cc / GBN, M4 / GBM), 256, 0, stream>>>(
      ybf, wbf_proj, x, x2, nullptr, M4, Cc, Cc, 0);

  // LN2
  ln_bf16_kernel<<<M4, 256, 0, stream>>>(x2, ln2_w, ln2_b, n2bf);

  // act = gelu(n2 @ w_fc) -> bf16 only (fp32 never materialized)
  gemm_bf16_kernel<<<dim3(HID / GBN, M4 / GBM), 256, 0, stream>>>(
      n2bf, wbf_fc, nullptr, nullptr, actbf, M4, HID, Cc, 1);

  // out = x2 + act @ w_mlp_proj -> fp32
  gemm_bf16_kernel<<<dim3(Cc / GBN, M4 / GBM), 256, 0, stream>>>(
      actbf, wbf_mlp, x2, out, nullptr, M4, Cc, HID, 0);
}
